// RPN_82154134438228
// MI455X (gfx1250) — compile-verified, matrix operand order checked by reading, not run on hardware
//
#include <hip/hip_runtime.h>

typedef __attribute__((ext_vector_type(16))) __bf16 bf16x16;
typedef __attribute__((ext_vector_type(8)))  float  f32x8;

union FragU { bf16x16 v; uint4 q[2]; };

__device__ __forceinline__ unsigned short f2bf(float x) {
    unsigned u = __float_as_uint(x);
    unsigned r = ((u >> 16) & 1u) + 0x7FFFu;
    return (unsigned short)((u + r) >> 16);
}
__device__ __forceinline__ f32x8 zero8() {
    f32x8 v = {0.f,0.f,0.f,0.f,0.f,0.f,0.f,0.f};
    return v;
}
__device__ __forceinline__ unsigned key_of(float f) {
    unsigned u = __float_as_uint(f);
    return (u & 0x80000000u) ? ~u : (u | 0x80000000u);
}

#define KEY_NEGINF 0x007FFFFFu
#define SUPW 96
#define CAND 4096

// ---------------------------------------------------------------------------
// Weight packing: conv_w (f32 OIHW 256x256x3x3) -> bf16 [tap][kc][m][32]
// (64B rows matching the 16-bit WMMA A lane layout), and cls/reg 1x1 weights
// into one 32x256 bf16 A2 matrix (rows 0..5 = cls, 16..27 = reg, rest 0).
// ---------------------------------------------------------------------------
__global__ void pack_kernel(const float* __restrict__ conv_w,
                            const float* __restrict__ cls_w,
                            const float* __restrict__ reg_w,
                            unsigned short* __restrict__ Wpk,
                            unsigned short* __restrict__ A2pk) {
    int i = blockIdx.x * blockDim.x + threadIdx.x;
    if (i < 589824) {
        int j = i & 31; int rest = i >> 5;
        int m = rest & 255; int tk = rest >> 8;      // tk = tap*8+kc, 0..71
        int kc = tk & 7, tap = tk >> 3;
        int k = kc * 32 + j;
        int ty = tap / 3, tx = tap % 3;
        float v = conv_w[((size_t)(m * 256 + k) * 3 + ty) * 3 + tx];
        Wpk[i] = f2bf(v);
    } else if (i < 589824 + 8192) {
        int t = i - 589824;
        int j = t & 31; int rest = t >> 5;
        int m = rest & 31; int kk = rest >> 5;       // kk = 0..7
        int k = kk * 32 + j;
        float v = 0.f;
        if (m < 6)                 v = cls_w[m * 256 + k];
        else if (m >= 16 && m < 28) v = reg_w[(m - 16) * 256 + k];
        A2pk[t] = f2bf(v);
    }
}

// ---------------------------------------------------------------------------
// Fused implicit-GEMM conv3x3(256->256)+ReLU + 1x1 heads via WMMA bf16.
// Block = 256 threads (8 waves). Tile = 64 pixels x 256 out channels.
// Wave w handles M rows [32w,32w+32) as 2 M-tiles x 4 N-tiles (pixels).
// Double-buffered LDS slab (one barrier per K-step), software-pipelined
// A-fragment (weight) loads, coalesced activation loads (wave = 4 channels
// x 32 consecutive pixels), distributed weight prefetch 4 steps ahead.
// ---------------------------------------------------------------------------
__global__ __launch_bounds__(256) void conv_wmma_kernel(
    const float* __restrict__ x,          // [2,256,H,W]
    const unsigned short* __restrict__ Wpk,
    const float* __restrict__ convb,
    const unsigned short* __restrict__ A2pk,
    const float* __restrict__ clsb,
    const float* __restrict__ regb,
    float* __restrict__ regout,           // [2, HW*3, 4]
    float* __restrict__ clsout,           // [2, HW*3, 2]
    int H, int W) {
    const int HW = H * W;
    const int nA = HW * 3;
    const int tiles = (HW + 63) / 64;
    const int b  = blockIdx.x / tiles;
    const int p0 = (blockIdx.x % tiles) * 64;
    const int tid = threadIdx.x;
    const int wv = tid >> 5, lane = tid & 31;
    const int l16 = lane & 15;
    const bool hi = lane >= 16;
    const int off = hi ? 8 : 0;

    __shared__ __align__(16) unsigned short slab[2][64 * 40];  // 64 pix x 32 k (+pad)
    __shared__ __align__(16) unsigned short hlds[64 * 264];    // 64 pix x 256 ch (+pad)

    f32x8 acc[2][4];
#pragma unroll
    for (int i = 0; i < 2; i++)
#pragma unroll
        for (int j = 0; j < 4; j++) acc[i][j] = zero8();

    // loader coords: lane -> 2 pixels (lane, lane+32); wave -> 4 consecutive ch
    const int pA = p0 + lane, pB = p0 + lane + 32;
    const int yA = pA / W, xA = pA % W;
    const int yB = pB / W, xB = pB % W;
    const int cbase = wv * 4;
    const float* xb = x + (size_t)b * 256 * HW;

    auto loadslab = [&](int tap, int kc, int buf) {
        const int dy = tap / 3 - 1, dx = tap % 3 - 1;
        const int syA = yA + dy, sxA = xA + dx;
        const int syB = yB + dy, sxB = xB + dx;
        const bool vA = (pA < HW) && (syA >= 0) && (syA < H) && (sxA >= 0) && (sxA < W);
        const bool vB = (pB < HW) && (syB >= 0) && (syB < H) && (sxB >= 0) && (sxB < W);
        const size_t sA = (size_t)(syA * W + sxA);
        const size_t sB = (size_t)(syB * W + sxB);
        const int c = kc * 32 + cbase;
        float a0 = 0.f, a1 = 0.f, a2 = 0.f, a3 = 0.f;
        float b0 = 0.f, b1 = 0.f, b2 = 0.f, b3 = 0.f;
        if (vA) {
            a0 = xb[(size_t)(c + 0) * HW + sA];
            a1 = xb[(size_t)(c + 1) * HW + sA];
            a2 = xb[(size_t)(c + 2) * HW + sA];
            a3 = xb[(size_t)(c + 3) * HW + sA];
        }
        if (vB) {
            b0 = xb[(size_t)(c + 0) * HW + sB];
            b1 = xb[(size_t)(c + 1) * HW + sB];
            b2 = xb[(size_t)(c + 2) * HW + sB];
            b3 = xb[(size_t)(c + 3) * HW + sB];
        }
        uint2 uA, uB;
        uA.x = (unsigned)f2bf(a0) | ((unsigned)f2bf(a1) << 16);
        uA.y = (unsigned)f2bf(a2) | ((unsigned)f2bf(a3) << 16);
        uB.x = (unsigned)f2bf(b0) | ((unsigned)f2bf(b1) << 16);
        uB.y = (unsigned)f2bf(b2) | ((unsigned)f2bf(b3) << 16);
        *(uint2*)&slab[buf][lane * 40 + cbase]        = uA;
        *(uint2*)&slab[buf][(lane + 32) * 40 + cbase] = uB;
    };

    auto loadA = [&](int it, FragU& f0, FragU& f1) {
        const unsigned short* wrow = Wpk + (size_t)it * 256 * 32;
        const unsigned short* q0 = wrow + (size_t)((wv * 2 + 0) * 16 + l16) * 32 + off;
        f0.q[0] = *(const uint4*)q0; f0.q[1] = *(const uint4*)(q0 + 16);
        const unsigned short* q1 = wrow + (size_t)((wv * 2 + 1) * 16 + l16) * 32 + off;
        f1.q[0] = *(const uint4*)q1; f1.q[1] = *(const uint4*)(q1 + 16);
    };

    FragU a0, a1, an0, an1;
    loadA(0, a0, a1);
    loadslab(0, 0, 0);
    __syncthreads();

    int cur = 0;
    for (int it = 0; it < 72; ++it) {
        const int nit = it + 1;
        if (nit < 72) {
            loadslab(nit >> 3, nit & 7, cur ^ 1);   // fill other buffer
            loadA(nit, an0, an1);                   // pipeline weights
        }
        const int pf = it + 4;
        if (pf < 72)                                 // distributed weight prefetch
            __builtin_prefetch((const void*)(Wpk + (size_t)pf * 256 * 32 + (size_t)tid * 32), 0, 1);

        FragU bf;
#pragma unroll
        for (int nt = 0; nt < 4; nt++) {
            const unsigned short* pBp = &slab[cur][(nt * 16 + l16) * 40 + off];
            bf.q[0] = *(const uint4*)pBp; bf.q[1] = *(const uint4*)(pBp + 16);
            acc[0][nt] = __builtin_amdgcn_wmma_f32_16x16x32_bf16(false, a0.v, false, bf.v, (short)0, acc[0][nt], false, false);
            acc[1][nt] = __builtin_amdgcn_wmma_f32_16x16x32_bf16(false, a1.v, false, bf.v, (short)0, acc[1][nt], false, false);
        }
        __syncthreads();
        a0 = an0; a1 = an1;
        cur ^= 1;
    }

    // epilogue 1: bias + ReLU, write h tile to LDS as bf16 (pixel-major, k contiguous)
#pragma unroll
    for (int mt = 0; mt < 2; mt++) {
        const int mbase = wv * 32 + mt * 16 + off;
#pragma unroll
        for (int nt = 0; nt < 4; nt++) {
            const int ph = nt * 16 + l16;
            unsigned short hv[8];
#pragma unroll
            for (int r = 0; r < 8; r++) {
                float h = acc[mt][nt][r] + convb[mbase + r];
                hv[r] = f2bf(fmaxf(h, 0.f));
            }
            uint4 qv;
            qv.x = (unsigned)hv[0] | ((unsigned)hv[1] << 16);
            qv.y = (unsigned)hv[2] | ((unsigned)hv[3] << 16);
            qv.z = (unsigned)hv[4] | ((unsigned)hv[5] << 16);
            qv.w = (unsigned)hv[6] | ((unsigned)hv[7] << 16);
            *(uint4*)&hlds[ph * 264 + mbase] = qv;
        }
    }
    __syncthreads();

    // stage 2: [32 x 256] x [256 x 64pix] WMMA GEMM for cls(6)+reg(12) heads
    {
        const int mt2 = wv & 1, nt2 = wv >> 1;   // 2 M-tiles x 4 N-tiles = 8 waves
        f32x8 acc2 = zero8();
#pragma unroll
        for (int kk = 0; kk < 8; kk++) {
            FragU a2, b2;
            const unsigned short* pAp = A2pk + (size_t)(kk * 32 + mt2 * 16 + l16) * 32 + off;
            a2.q[0] = *(const uint4*)pAp; a2.q[1] = *(const uint4*)(pAp + 16);
            const unsigned short* pBp = &hlds[(nt2 * 16 + l16) * 264 + kk * 32 + off];
            b2.q[0] = *(const uint4*)pBp; b2.q[1] = *(const uint4*)(pBp + 16);
            acc2 = __builtin_amdgcn_wmma_f32_16x16x32_bf16(false, a2.v, false, b2.v, (short)0, acc2, false, false);
        }
        const int pix = p0 + nt2 * 16 + l16;
        if (pix < HW) {
            if (mt2 == 0) {
                if (!hi) {  // rows 0..7 -> cls channels 0..5
#pragma unroll
                    for (int r = 0; r < 6; r++)
                        clsout[(size_t)b * nA * 2 + (size_t)pix * 6 + r] = acc2[r] + clsb[r];
                }
            } else {        // rows 16..23 -> reg 0..7 ; rows 24..31 -> reg 8..11
                const int base = hi ? 8 : 0;
                const int lim  = hi ? 4 : 8;
#pragma unroll
                for (int r = 0; r < 8; r++) {
                    if (r < lim) {
                        const int rc = base + r;
                        regout[(size_t)b * nA * 4 + (size_t)pix * 12 + rc] = acc2[r] + regb[rc];
                    }
                }
            }
        }
    }
}

// ---------------------------------------------------------------------------
// Score + box decode: softmax fg score, anchor decode, clip, min-size filter.
// ---------------------------------------------------------------------------
__global__ void decode_kernel(const float* __restrict__ cls,
                              const float* __restrict__ reg,
                              float* __restrict__ score,
                              float* __restrict__ boxb,
                              int nA, int W, float stride, float minsize) {
    int i = blockIdx.x * blockDim.x + threadIdx.x;
    if (i >= 2 * nA) return;
    int b = i / nA, r = i % nA;
    int pix = r / 3, a = r % 3;
    float c0 = cls[((size_t)b * nA + r) * 2 + 0];
    float c1 = cls[((size_t)b * nA + r) * 2 + 1];
    float s = 1.f / (1.f + expf(c0 - c1));
    const float* rp = reg + ((size_t)b * nA + r) * 4;
    const float sr[3] = {0.70710678118654752f, 1.f, 1.41421356237309515f};
    float size = stride * 8.f;
    float aw = size * sr[a], ah = size / sr[a];
    float acx = ((float)(pix % W) + 0.5f) * stride;
    float acy = ((float)(pix / W) + 0.5f) * stride;
    float cx = rp[0] * aw + acx, cy = rp[1] * ah + acy;
    float w2 = expf(rp[2]) * aw, h2 = expf(rp[3]) * ah;
    float x1 = cx - 0.5f * w2, y1 = cy - 0.5f * h2;
    float x2 = cx + 0.5f * w2, y2 = cy + 0.5f * h2;
    x1 = fminf(fmaxf(x1, 0.f), 768.f); y1 = fminf(fmaxf(y1, 0.f), 768.f);
    x2 = fminf(fmaxf(x2, 0.f), 768.f); y2 = fminf(fmaxf(y2, 0.f), 768.f);
    float bw = x2 - x1, bh = y2 - y1;
    if (!(bw >= minsize && bh >= minsize)) s = -__builtin_inff();
    score[(size_t)b * nA + r] = s;
    float* bo = boxb + ((size_t)b * nA + r) * 4;
    bo[0] = x1; bo[1] = y1; bo[2] = x2; bo[3] = y2;
}

// ---------------------------------------------------------------------------
// Top-K (pre-NMS) via 1024-bucket radix histogram + gather + bitonic sort
// ---------------------------------------------------------------------------
__global__ void hist_kernel(const float* __restrict__ score, int n,
                            unsigned* __restrict__ hist) {
    int i = blockIdx.x * blockDim.x + threadIdx.x;
    if (i >= n) return;
    atomicAdd(&hist[key_of(score[i]) >> 22], 1u);
}

__global__ void thresh_kernel(const unsigned* __restrict__ hist, int K,
                              unsigned* __restrict__ thr) {
    unsigned acc = 0, th = 0;
    for (int t = 1023; t >= 0; t--) {
        acc += hist[t];
        if (acc >= (unsigned)K) { th = (unsigned)t << 22; break; }
    }
    *thr = th;
}

__global__ void gather_kernel(const float* __restrict__ score, int n,
                              const unsigned* __restrict__ thr,
                              unsigned* __restrict__ ckey, unsigned* __restrict__ cidx,
                              unsigned* __restrict__ counter) {
    int i = blockIdx.x * blockDim.x + threadIdx.x;
    if (i >= n) return;
    unsigned k = key_of(score[i]);
    if (k >= *thr) {
        unsigned p = atomicAdd(counter, 1u);
        if (p < CAND) { ckey[p] = k; cidx[p] = (unsigned)i; }
    }
}

__global__ __launch_bounds__(1024) void sort_gather_kernel(
    const unsigned* __restrict__ ckey, const unsigned* __restrict__ cidx,
    const unsigned* __restrict__ counter, int K,
    const float* __restrict__ boxsrc,
    unsigned* __restrict__ selkey, unsigned* __restrict__ selidx,
    float* __restrict__ nmsbox) {
    __shared__ unsigned long long s[CAND];
    int tid = threadIdx.x;
    unsigned cnt = *counter; if (cnt > CAND) cnt = CAND;
    for (int i = tid; i < CAND; i += 1024) {
        unsigned k  = (i < (int)cnt) ? ckey[i] : 0u;
        unsigned id = (i < (int)cnt) ? cidx[i] : 0xFFFFFFFFu;
        s[i] = ((unsigned long long)k << 32) | (unsigned)(~id);
    }
    __syncthreads();
    for (int k2 = 2; k2 <= CAND; k2 <<= 1)
        for (int j = k2 >> 1; j > 0; j >>= 1) {
            for (int i = tid; i < CAND; i += 1024) {
                int ixj = i ^ j;
                if (ixj > i) {
                    unsigned long long a = s[i], bb = s[ixj];
                    bool sw = ((i & k2) == 0) ? (a < bb) : (a > bb);  // descending
                    if (sw) { s[i] = bb; s[ixj] = a; }
                }
            }
            __syncthreads();
        }
    for (int i = tid; i < K; i += 1024) {
        unsigned long long v = s[i];
        unsigned k  = (unsigned)(v >> 32);
        unsigned id = ~((unsigned)v);
        selkey[i] = k; selidx[i] = id;
        float b0 = 0, b1 = 0, b2 = 0, b3 = 0;
        if (id != 0xFFFFFFFFu) {
            const float* bp = boxsrc + (size_t)id * 4;
            b0 = bp[0]; b1 = bp[1]; b2 = bp[2]; b3 = bp[3];
        }
        nmsbox[(size_t)i * 4 + 0] = b0; nmsbox[(size_t)i * 4 + 1] = b1;
        nmsbox[(size_t)i * 4 + 2] = b2; nmsbox[(size_t)i * 4 + 3] = b3;
    }
}

// ---------------------------------------------------------------------------
// NMS suppression matrix (bit per pair, j>i) + sequential keep reduction
// ---------------------------------------------------------------------------
__global__ void iou_kernel(const float* __restrict__ box, int K,
                           unsigned* __restrict__ sup) {
    int i = blockIdx.x;
    float x1 = box[(size_t)i * 4 + 0], y1 = box[(size_t)i * 4 + 1];
    float x2 = box[(size_t)i * 4 + 2], y2 = box[(size_t)i * 4 + 3];
    float ai = (x2 - x1) * (y2 - y1);
    int wdi = threadIdx.x;
    unsigned m = 0;
    for (int t = 0; t < 32; t++) {
        int j = wdi * 32 + t;
        if (j > i && j < K) {
            float bx1 = box[(size_t)j * 4 + 0], by1 = box[(size_t)j * 4 + 1];
            float bx2 = box[(size_t)j * 4 + 2], by2 = box[(size_t)j * 4 + 3];
            float ix1 = fmaxf(x1, bx1), iy1 = fmaxf(y1, by1);
            float ix2 = fminf(x2, bx2), iy2 = fminf(y2, by2);
            float inter = fmaxf(ix2 - ix1, 0.f) * fmaxf(iy2 - iy1, 0.f);
            float aj = (bx2 - bx1) * (by2 - by1);
            float iou = inter / (ai + aj - inter + 1e-9f);
            if (iou > 0.7f) m |= 1u << t;
        }
    }
    sup[(size_t)i * SUPW + wdi] = m;
}

__global__ __launch_bounds__(128) void finalize_kernel(
    const unsigned* __restrict__ selkey, const unsigned* __restrict__ selidx,
    const unsigned* __restrict__ sup, const float* __restrict__ nmsbox,
    int K, int W, float stride,
    float* __restrict__ roi, float* __restrict__ org, float* __restrict__ idxo,
    int b) {
    __shared__ unsigned keep[SUPW];
    int tid = threadIdx.x;
    int nw = (K + 31) / 32;
    for (int w = tid; w < SUPW; w += blockDim.x) {
        unsigned m = 0xFFFFFFFFu;
        int rem = K - w * 32;
        if (rem <= 0) m = 0; else if (rem < 32) m = (1u << rem) - 1u;
        keep[w] = m;
    }
    __syncthreads();
    for (int i = 0; i < K; i++) {
        unsigned alive = (keep[i >> 5] >> (i & 31)) & 1u;
        __syncthreads();
        if (alive) {
            const unsigned* srow = sup + (size_t)i * SUPW;
            for (int w = tid; w < nw; w += blockDim.x) keep[w] &= ~srow[w];
        }
        __syncthreads();
    }
    if (tid == 0) {
        int pos[100];
        int cnt = 0;
        for (int i = 0; i < K && cnt < 100; i++)
            if (((keep[i >> 5] >> (i & 31)) & 1u) && selkey[i] > KEY_NEGINF) pos[cnt++] = i;
        for (int i = 0; i < K && cnt < 100; i++) {
            bool used = ((keep[i >> 5] >> (i & 31)) & 1u) && selkey[i] > KEY_NEGINF;
            if (!used) pos[cnt++] = i;
        }
        while (cnt < 100) pos[cnt++] = 0;
        const float sr[3] = {0.70710678118654752f, 1.f, 1.41421356237309515f};
        for (int j = 0; j < 100; j++) {
            int p = pos[j];
            const float* bx = nmsbox + (size_t)p * 4;
            size_t ro = ((size_t)b * 100 + j) * 4;
            roi[ro + 0] = bx[0]; roi[ro + 1] = bx[1];
            roi[ro + 2] = bx[2]; roi[ro + 3] = bx[3];
            unsigned oi = selidx[p]; if (oi == 0xFFFFFFFFu) oi = 0;
            int pix = (int)(oi / 3), a = (int)(oi % 3);
            float size = stride * 8.f;
            float aw = size * sr[a], ah = size / sr[a];
            float cx = ((float)(pix % W) + 0.5f) * stride;
            float cy = ((float)(pix / W) + 0.5f) * stride;
            org[ro + 0] = cx - 0.5f * aw; org[ro + 1] = cy - 0.5f * ah;
            org[ro + 2] = cx + 0.5f * aw; org[ro + 3] = cy + 0.5f * ah;
            idxo[(size_t)b * 100 + j] = (float)b;
        }
    }
}

// ---------------------------------------------------------------------------
extern "C" void kernel_launch(void* const* d_in, const int* in_sizes, int n_in,
                              void* d_out, int out_size, void* d_ws, size_t ws_size,
                              hipStream_t stream) {
    (void)in_sizes; (void)n_in; (void)out_size; (void)ws_size;
    const float* x[5];
    for (int i = 0; i < 5; i++) x[i] = (const float*)d_in[i];
    const float* conv_w = (const float*)d_in[5];
    const float* conv_b = (const float*)d_in[6];
    const float* cls_w  = (const float*)d_in[7];
    const float* cls_b  = (const float*)d_in[8];
    const float* reg_w  = (const float*)d_in[9];
    const float* reg_b  = (const float*)d_in[10];
    float* out = (float*)d_out;
    char* ws = (char*)d_ws;

    static const int G[5] = {192, 96, 48, 24, 12};

    // d_out layout (flat, return order): regs[5], clss[5], rois[5], origins[5], idxs[5]
    size_t regOff[5], clsOff[5], roiOff[5], orgOff[5], idxOff[5];
    {
        size_t off = 0;
        for (int l = 0; l < 5; l++) { size_t nA = (size_t)G[l] * G[l] * 3; regOff[l] = off; off += 2 * nA * 4; }
        for (int l = 0; l < 5; l++) { size_t nA = (size_t)G[l] * G[l] * 3; clsOff[l] = off; off += 2 * nA * 2; }
        for (int l = 0; l < 5; l++) { roiOff[l] = off; off += 800; }
        for (int l = 0; l < 5; l++) { orgOff[l] = off; off += 800; }
        for (int l = 0; l < 5; l++) { idxOff[l] = off; off += 200; }
    }

    // workspace layout
    size_t wo = 0;
    auto A = [&](size_t bytes) { size_t r = wo; wo = (wo + bytes + 255) & ~(size_t)255; return r; };
    size_t WpkO = A(589824 * 2);
    size_t A2O  = A(8192 * 2);
    size_t scO[5], bxO[5];
    for (int l = 0; l < 5; l++) {
        size_t nA = (size_t)G[l] * G[l] * 3;
        scO[l] = A(2 * nA * 4);
        bxO[l] = A(2 * nA * 16);
    }
    size_t zeroO = A(10 * 4096 + 10 * 4);   // 10 hists (1024 u32) + 10 counters
    size_t thO   = A(10 * 4);
    size_t ckO   = A((size_t)10 * CAND * 4);
    size_t ciO   = A((size_t)10 * CAND * 4);
    size_t skO   = A((size_t)10 * CAND * 4);
    size_t siO   = A((size_t)10 * CAND * 4);
    size_t nbO   = A((size_t)10 * 3000 * 16);
    size_t supO  = A((size_t)10 * 3000 * SUPW * 4);

    hipMemsetAsync(ws + zeroO, 0, 10 * 4096 + 10 * 4, stream);

    pack_kernel<<<(589824 + 8192 + 255) / 256, 256, 0, stream>>>(
        conv_w, cls_w, reg_w, (unsigned short*)(ws + WpkO), (unsigned short*)(ws + A2O));

    for (int l = 0; l < 5; l++) {
        int g = G[l], HW = g * g;
        int tiles = (HW + 63) / 64;
        conv_wmma_kernel<<<2 * tiles, 256, 0, stream>>>(
            x[l], (const unsigned short*)(ws + WpkO), conv_b,
            (const unsigned short*)(ws + A2O), cls_b, reg_b,
            out + regOff[l], out + clsOff[l], g, g);
    }

    for (int l = 0; l < 5; l++) {
        int g = G[l]; int nA = g * g * 3;
        float sl = (float)(4 << l);
        decode_kernel<<<(2 * nA + 255) / 256, 256, 0, stream>>>(
            out + clsOff[l], out + regOff[l],
            (float*)(ws + scO[l]), (float*)(ws + bxO[l]), nA, g, sl, sl);
    }

    for (int l = 0; l < 5; l++) {
        int g = G[l]; int nA = g * g * 3;
        int K = nA < 3000 ? nA : 3000;
        float sl = (float)(4 << l);
        for (int b = 0; b < 2; b++) {
            int q = l * 2 + b;
            const float* sc = (const float*)(ws + scO[l]) + (size_t)b * nA;
            const float* bx = (const float*)(ws + bxO[l]) + (size_t)b * nA * 4;
            unsigned* hist = (unsigned*)(ws + zeroO) + (size_t)q * 1024;
            unsigned* ctr  = (unsigned*)(ws + zeroO + 10 * 4096) + q;
            unsigned* thr  = (unsigned*)(ws + thO) + q;
            unsigned* ck   = (unsigned*)(ws + ckO) + (size_t)q * CAND;
            unsigned* ci   = (unsigned*)(ws + ciO) + (size_t)q * CAND;
            unsigned* sk   = (unsigned*)(ws + skO) + (size_t)q * CAND;
            unsigned* si   = (unsigned*)(ws + siO) + (size_t)q * CAND;
            float*    nb   = (float*)(ws + nbO) + (size_t)q * 3000 * 4;
            unsigned* sp   = (unsigned*)(ws + supO) + (size_t)q * 3000 * SUPW;

            hist_kernel<<<(nA + 255) / 256, 256, 0, stream>>>(sc, nA, hist);
            thresh_kernel<<<1, 1, 0, stream>>>(hist, K, thr);
            gather_kernel<<<(nA + 255) / 256, 256, 0, stream>>>(sc, nA, thr, ck, ci, ctr);
            sort_gather_kernel<<<1, 1024, 0, stream>>>(ck, ci, ctr, K, bx, sk, si, nb);
            iou_kernel<<<K, SUPW, 0, stream>>>(nb, K, sp);
            finalize_kernel<<<1, 128, 0, stream>>>(sk, si, sp, nb, K, g, sl,
                                                   out + roiOff[l], out + orgOff[l],
                                                   out + idxOff[l], b);
        }
    }
}
